// LatticeIsingModel_7945689498216
// MI455X (gfx1250) — 8-bit matrix operand order repaired and checked by reading, compiled, not run on hardware
//
#include <hip/hip_runtime.h>

typedef __attribute__((ext_vector_type(8))) int   v8i;
typedef __attribute__((ext_vector_type(8))) float v8f;

#define NN      10000
#define BATCH   512
#define BN      64
#define KT      64
#define LDSB    80
#define THREADS 512
#define KSPLIT  4
#define KSTEPS_PER_SPLIT 40

union FragI { uint2 d[4]; uint4 q[2]; v8i v; };

__device__ __forceinline__ unsigned spin8(float x01) {
    return x01 > 0.5f ? 0x38u : 0xB8u;
}
__device__ __forceinline__ unsigned adj8(float g01) {
    return g01 > 0.5f ? 0x38u : 0x00u;
}

__global__ __launch_bounds__(256) void ising_bias_kernel(
    const float* __restrict__ x, const float* __restrict__ bias,
    float* __restrict__ out)
{
    __shared__ float red[256];
    const int b = blockIdx.x;
    const int t = threadIdx.x;
    float s = 0.f;
    const float* xr = x + (size_t)b * NN;
    for (int i = t; i < NN; i += 256)
        s += (2.f * xr[i] - 1.f) * bias[i];
    red[t] = s;
    __syncthreads();
    for (int w = 128; w > 0; w >>= 1) {
        if (t < w) red[t] += red[t + w];
        __syncthreads();
    }
    if (t == 0) out[b] = red[0];
}

__global__ __launch_bounds__(THREADS) void ising_wmma_kernel(
    const float* __restrict__ x, const float* __restrict__ G,
    const float* __restrict__ sigmaP, float* __restrict__ out)
{
    __shared__ __align__(16) unsigned char As8[BATCH * LDSB];
    __shared__ __align__(16) unsigned char Bs8[BN * LDSB];
    unsigned int* As32 = (unsigned int*)As8;

    const int t    = threadIdx.x;
    const int lane = t & 31;
    const int wave = t >> 5;
    const int ns   = wave & 3;
    const int mg   = wave >> 2;
    const int lr   = lane & 15;
    const int hsel = (lane < 16) ? 0 : 1;
    const int n0   = blockIdx.x * BN;

    v8f acc[8];
    #pragma unroll
    for (int i = 0; i < 8; ++i) acc[i] = (v8f){0.f,0.f,0.f,0.f,0.f,0.f,0.f,0.f};

    for (int ks = 0; ks < KSTEPS_PER_SPLIT; ++ks) {
        const int k0 = (blockIdx.y * KSTEPS_PER_SPLIT + ks) * KT;
        if (k0 >= NN) break;

        if (k0 + KT <= NN) {
            #pragma unroll 4
            for (int p = 0; p < 16; ++p) {
                const int q   = p * THREADS + t;
                const int row = q >> 4;
                const int qc  = q & 15;
                const float4 v = *(const float4*)(x + (size_t)row * NN + k0 + qc * 4);
                unsigned u = spin8(v.x) | (spin8(v.y) << 8) |
                             (spin8(v.z) << 16) | (spin8(v.w) << 24);
                As32[row * (LDSB / 4) + qc] = u;
            }
        } else {
            for (int p = 0; p < 64; ++p) {
                const int e   = p * THREADS + t;
                const int row = e >> 6;
                const int col = e & 63;
                const int k   = k0 + col;
                unsigned b = (k < NN) ? spin8(x[(size_t)row * NN + k]) : 0u;
                As8[row * LDSB + col] = (unsigned char)b;
            }
        }
        #pragma unroll
        for (int p = 0; p < 8; ++p) {
            const int e  = p * THREADS + t;
            const int kk = e >> 6;
            const int nn = e & 63;
            const int k  = k0 + kk;
            const int n  = n0 + nn;
            const bool ok = (k < NN) && (n < NN);
            const float g = ok ? G[(size_t)k * NN + n] : 0.f;
            Bs8[nn * LDSB + kk] = (unsigned char)(ok ? adj8(g) : 0u);
            if (p == 0 && ok && (k0 + KT) < NN)
                __builtin_prefetch(&G[(size_t)(k + KT) * NN + n], 0, 1);
        }
        __syncthreads();

        FragI fb;
        {
            const unsigned char* bp = Bs8 + (ns * 16 + lr) * LDSB + hsel * 8;
            fb.d[0] = *(const uint2*)(bp);
            fb.d[1] = *(const uint2*)(bp + 16);
            fb.d[2] = *(const uint2*)(bp + 32);
            fb.d[3] = *(const uint2*)(bp + 48);
        }
        #pragma unroll
        for (int g4 = 0; g4 < 2; ++g4) {
            FragI fa[4];
            #pragma unroll
            for (int i = 0; i < 4; ++i) {
                const int row = (mg * 8 + g4 * 4 + i) * 16 + lr;
                const unsigned char* ap = As8 + row * LDSB + hsel * 8;
                fa[i].d[0] = *(const uint2*)(ap);
                fa[i].d[1] = *(const uint2*)(ap + 16);
                fa[i].d[2] = *(const uint2*)(ap + 32);
                fa[i].d[3] = *(const uint2*)(ap + 48);
            }
            #pragma unroll
            for (int i = 0; i < 4; ++i)
                acc[g4 * 4 + i] = __builtin_amdgcn_wmma_f32_16x16x64_fp8_fp8(
                    fa[i].v, fb.v, (short)0, acc[g4 * 4 + i], false, false);
        }
        __syncthreads();
    }

    const float sigma = *sigmaP;
    const int n = n0 + ns * 16 + lr;
    #pragma unroll
    for (int mt = 0; mt < 8; ++mt) {
        const int bRow0 = (mg * 8 + mt) * 16;
        #pragma unroll
        for (int r = 0; r < 8; ++r) {
            const int bIdx = bRow0 + r + hsel * 8;
            float val = acc[mt][r];
            const float xv = (n < NN) ? x[(size_t)bIdx * NN + n] : 0.5f;
            val *= (2.f * xv - 1.f);
            val += __shfl_xor(val, 1);
            val += __shfl_xor(val, 2);
            val += __shfl_xor(val, 4);
            val += __shfl_xor(val, 8);
            if (lr == 0) atomicAdd(&out[bIdx], sigma * val);
        }
    }
}

extern "C" void kernel_launch(void* const* d_in, const int* in_sizes, int n_in,
                              void* d_out, int out_size, void* d_ws, size_t ws_size,
                              hipStream_t stream) {
    const float* x     = (const float*)d_in[0];
    const float* G     = (const float*)d_in[1];
    const float* sigma = (const float*)d_in[2];
    const float* bias  = (const float*)d_in[3];
    float* out = (float*)d_out;

    ising_bias_kernel<<<BATCH, 256, 0, stream>>>(x, bias, out);

    dim3 grid((NN + BN - 1) / BN, KSPLIT);
    ising_wmma_kernel<<<grid, THREADS, 0, stream>>>(x, G, sigma, out);
}
